// AutoEncoder_66632122630651
// MI455X (gfx1250) — compile-verified, matrix operand order checked by reading, not run on hardware
//
#include <hip/hip_runtime.h>
#include <hip/hip_bf16.h>

// ---------------------------------------------------------------------------
// Problem sizes (fixed by the reference)
// ---------------------------------------------------------------------------
#define BATCH   4096
#define EMBED   2048
#define NFEAT   32768
#define TOPK    64

// GEMM tiling
#define BM 128
#define BN 128
#define BK 32

typedef __attribute__((ext_vector_type(16))) __bf16    bf16x16;
typedef __attribute__((ext_vector_type(8)))  float     v8f;
typedef __attribute__((ext_vector_type(4)))  unsigned  u32x4;

// ---------------------------------------------------------------------------
// Helpers
// ---------------------------------------------------------------------------
__device__ __forceinline__ unsigned short f32_to_bf16_rne(float f) {
    unsigned u = __float_as_uint(f);
    unsigned lsb = (u >> 16) & 1u;
    u += 0x7FFFu + lsb;
    return (unsigned short)(u >> 16);
}

// order-preserving float<->uint transforms (for radix select)
__device__ __forceinline__ unsigned f2ord(float f) {
    unsigned b = __float_as_uint(f);
    return b ^ ((b & 0x80000000u) ? 0xFFFFFFFFu : 0x80000000u);
}
__device__ __forceinline__ float ord2f(unsigned u) {
    unsigned b = (u & 0x80000000u) ? (u ^ 0x80000000u) : ~u;
    return __uint_as_float(b);
}

// CDNA5 async global->LDS copy (ASYNCcnt-tracked, no VGPR staging).
// ldsOff is the wave-relative LDS byte offset (low 32 bits of the generic
// pointer to a __shared__ object per the flat-aperture mapping).
__device__ __forceinline__ void async_copy_b128(const void* gptr, unsigned ldsOff) {
    asm volatile("global_load_async_to_lds_b128 %0, %1, off"
                 :: "v"(ldsOff), "v"(gptr) : "memory");
}
__device__ __forceinline__ unsigned lds_offset(const void* p) {
    return (unsigned)(unsigned long long)p;   // addr[31:0] == LDS offset
}

// Load a 16xBK bf16 A/B fragment from a row-major LDS tile (row stride = BK
// elements). Follows the CDNA5 16-bit A-matrix layout: lanes 0-15 hold
// K {0..7,16..23}, lanes 16-31 hold K {8..15,24..31} -> two b128 LDS loads.
__device__ __forceinline__ bf16x16 load_frag(const unsigned short* base, int lane) {
    int m = lane & 15;
    int h = lane >> 4;
    const unsigned short* p = base + m * BK + h * 8;
    union { bf16x16 v; u32x4 q[2]; } f;
    f.q[0] = *(const u32x4*)(p);        // bytes +0  (h=0: K0..7  ; h=1: K8..15)
    f.q[1] = *(const u32x4*)(p + 16);   // bytes +32 (h=0: K16..23; h=1: K24..31)
    return f.v;
}

// ---------------------------------------------------------------------------
// 0) zero the feature flags
// ---------------------------------------------------------------------------
__global__ void zero_flags_kernel(int* __restrict__ flags) {
    int i = blockIdx.x * blockDim.x + threadIdx.x;
    if (i < NFEAT) flags[i] = 0;
}

// ---------------------------------------------------------------------------
// 1) X = bf16(embed - bias)   [BATCH x EMBED]
// ---------------------------------------------------------------------------
__global__ void convert_x_kernel(const float* __restrict__ embed,
                                 const float* __restrict__ bias,
                                 unsigned short* __restrict__ Xb) {
    int i = blockIdx.x * blockDim.x + threadIdx.x;
    if (i < BATCH * EMBED) {
        float v = embed[i] - bias[i & (EMBED - 1)];
        Xb[i] = f32_to_bf16_rne(v);
    }
}

// ---------------------------------------------------------------------------
// 2) Wb = bf16(enc_weight)    [NFEAT x EMBED]
// ---------------------------------------------------------------------------
__global__ void convert_w_kernel(const float* __restrict__ W,
                                 unsigned short* __restrict__ Wb) {
    long long i = (long long)blockIdx.x * blockDim.x + threadIdx.x;
    if (i < (long long)NFEAT * EMBED) {
        Wb[i] = f32_to_bf16_rne(W[i]);
    }
}

// ---------------------------------------------------------------------------
// 3) Projection GEMM: P[b,f] = sum_k X[b,k] * W[f,k]   (bf16 in, f32 out)
//    grid = (NFEAT/BN, BATCH/BM), block = 256 (8 waves, 2x4 wave grid).
//    Double-buffered: tile t+1 streams memory->LDS via async-to-LDS copies
//    (ASYNCcnt) while tile t is consumed by WMMA.
// ---------------------------------------------------------------------------
__global__ __launch_bounds__(256) void gemm_bf16_kernel(
        const unsigned short* __restrict__ X,   // [BATCH x EMBED] bf16
        const unsigned short* __restrict__ W,   // [NFEAT x EMBED] bf16
        float* __restrict__ P)                  // [BATCH x NFEAT] f32
{
    __shared__ unsigned short sA[2][BM * BK];   // 2 x 8 KB
    __shared__ unsigned short sB[2][BN * BK];   // 2 x 8 KB

    const int tid  = threadIdx.x;
    const int lane = tid & 31;
    const int wave = tid >> 5;
    const int wm   = wave & 1;     // 2 waves along M  -> 64 rows each
    const int wn   = wave >> 1;    // 4 waves along N  -> 32 cols each

    const int rowBlk = blockIdx.y * BM;
    const int colBlk = blockIdx.x * BN;

    v8f acc[4][2];
#pragma unroll
    for (int i = 0; i < 4; ++i)
#pragma unroll
        for (int j = 0; j < 2; ++j)
            acc[i][j] = (v8f)(0.0f);

    const int r = tid >> 1;              // 0..127 (tile row loaded by this thread)
    const int c = (tid & 1) * 16;        // 0 or 16 (K offset)

    const unsigned short* gXrow = X + (rowBlk + r) * EMBED + c;
    const unsigned short* gWrow = W + (colBlk + r) * EMBED + c;
    const unsigned aOff0 = lds_offset(&sA[0][r * BK + c]);
    const unsigned aOff1 = lds_offset(&sA[1][r * BK + c]);
    const unsigned bOff0 = lds_offset(&sB[0][r * BK + c]);
    const unsigned bOff1 = lds_offset(&sB[1][r * BK + c]);

    // prologue: start tile 0
    async_copy_b128(gXrow,     aOff0);
    async_copy_b128(gXrow + 8, aOff0 + 16);
    async_copy_b128(gWrow,     bOff0);
    async_copy_b128(gWrow + 8, bOff0 + 16);

    const int T = EMBED / BK;            // 64 K-tiles
    for (int t = 0; t < T; ++t) {
        const int cur = t & 1;
        if (t + 1 < T) {
            // stream next tile into the other buffer while we compute
            const int k1 = (t + 1) * BK;
            const unsigned aOff = cur ? aOff0 : aOff1;
            const unsigned bOff = cur ? bOff0 : bOff1;
            async_copy_b128(gXrow + k1,     aOff);
            async_copy_b128(gXrow + k1 + 8, aOff + 16);
            async_copy_b128(gWrow + k1,     bOff);
            async_copy_b128(gWrow + k1 + 8, bOff + 16);
            if (t + 2 < T) {              // GL2 prefetch two tiles ahead
                __builtin_prefetch(gXrow + k1 + BK, 0, 1);
                __builtin_prefetch(gWrow + k1 + BK, 0, 1);
            }
            // wait for tile t's 4 copies (in-order); leave next 4 in flight
            asm volatile("s_wait_asynccnt 0x4" ::: "memory");
        } else {
            asm volatile("s_wait_asynccnt 0x0" ::: "memory");
        }
        __syncthreads();

        bf16x16 aF[4], bF[2];
#pragma unroll
        for (int i = 0; i < 4; ++i)
            aF[i] = load_frag(&sA[cur][(wm * 64 + i * 16) * BK], lane);
#pragma unroll
        for (int j = 0; j < 2; ++j)
            bF[j] = load_frag(&sB[cur][(wn * 32 + j * 16) * BK], lane);

#pragma unroll
        for (int i = 0; i < 4; ++i)
#pragma unroll
            for (int j = 0; j < 2; ++j)
                acc[i][j] = __builtin_amdgcn_wmma_f32_16x16x32_bf16(
                    /*neg_a=*/false, aF[i], /*neg_b=*/false, bF[j],
                    /*c_mod=*/(short)0, acc[i][j],
                    /*reuse_a=*/false, /*reuse_b=*/false);

        __syncthreads();   // all waves done with `cur` before t+2 overwrites it
    }

    // C/D layout: VGPR rr -> lanes 0-15: M=rr, N=lane ; lanes 16-31: M=rr+8
    const int h = lane >> 4;
    const int n = lane & 15;
#pragma unroll
    for (int i = 0; i < 4; ++i) {
        const int mBase = rowBlk + wm * 64 + i * 16 + 8 * h;
#pragma unroll
        for (int j = 0; j < 2; ++j) {
            const int col = colBlk + wn * 32 + j * 16 + n;
#pragma unroll
            for (int rr = 0; rr < 8; ++rr) {
                P[(size_t)(mBase + rr) * NFEAT + col] = acc[i][j][rr];
            }
        }
    }
}

// ---------------------------------------------------------------------------
// 4) Radix top-64 per row. One 256-thread block per row.
//    Row cached in LDS as order-preserving uints (128 KB of the 320 KB WGP LDS).
// ---------------------------------------------------------------------------
__global__ __launch_bounds__(256) void topk_kernel(
        const float* __restrict__ P,
        int* __restrict__ feats,        // [BATCH x TOPK]
        float* __restrict__ wts,        // [BATCH x TOPK]
        int* __restrict__ flags)        // [NFEAT]
{
    __shared__ unsigned sRow[NFEAT];    // 128 KB
    __shared__ unsigned hist[256];
    __shared__ unsigned sPrefix, sKrem;
    __shared__ int cntG, cntE;

    const int row = blockIdx.x;
    const int tid = threadIdx.x;
    const float* pr = P + (size_t)row * NFEAT;

    for (int i = tid; i < NFEAT; i += 256) sRow[i] = f2ord(pr[i]);
    if (tid == 0) { sPrefix = 0u; sKrem = TOPK; cntG = 0; cntE = 0; }
    __syncthreads();

    // 4 radix-256 rounds, MSB first, to find the 64th-largest value T
    for (int d = 3; d >= 0; --d) {
        const int sh = d * 8;
        const unsigned maskPrev = (d == 3) ? 0u : (0xFFFFFFFFu << ((d + 1) * 8));
        hist[tid] = 0u;
        __syncthreads();
        const unsigned prefix = sPrefix;
        for (int i = tid; i < NFEAT; i += 256) {
            unsigned u = sRow[i];
            if ((u & maskPrev) == prefix)
                atomicAdd(&hist[(u >> sh) & 255u], 1u);
        }
        __syncthreads();
        if (tid == 0) {
            unsigned kr = sKrem;
            int bin = 255;
            while (bin > 0 && hist[bin] < kr) { kr -= hist[bin]; --bin; }
            sKrem   = kr;
            sPrefix = sPrefix | ((unsigned)bin << sh);
        }
        __syncthreads();
    }

    const unsigned T      = sPrefix;   // value of the 64th largest element
    const unsigned needEq = sKrem;     // how many == T make the cut
    const int base = row * TOPK;

    // strictly greater than T: always in top-k
    for (int i = tid; i < NFEAT; i += 256) {
        unsigned u = sRow[i];
        if (u > T) {
            int p = atomicAdd(&cntG, 1);
            feats[base + p] = i;
            wts[base + p]   = ord2f(u);
            flags[i] = 1;               // benign race: everyone writes 1
        }
    }
    __syncthreads();
    const int g = cntG;                 // == TOPK - needEq

    // fill remaining slots with elements equal to T
    for (int i = tid; i < NFEAT; i += 256) {
        unsigned u = sRow[i];
        if (u == T) {
            int p = atomicAdd(&cntE, 1);
            if (p < (int)needEq) {
                feats[base + g + p] = i;
                wts[base + g + p]   = ord2f(u);
                flags[i] = 1;
            }
        }
    }
}

// ---------------------------------------------------------------------------
// 5) Decoder: recon = sum_k w_k * lookup[feat_k] + bias, then L2-normalize.
//    One 256-thread block per row; each thread owns 8 output columns.
// ---------------------------------------------------------------------------
__global__ __launch_bounds__(256) void decode_norm_kernel(
        const int* __restrict__ feats,
        const float* __restrict__ wts,
        const float* __restrict__ lookup,   // [NFEAT x EMBED]
        const float* __restrict__ bias,     // [EMBED]
        float* __restrict__ out)            // [BATCH x EMBED]
{
    __shared__ int   sF[TOPK];
    __shared__ float sW[TOPK];
    __shared__ float red[256];

    const int row = blockIdx.x;
    const int tid = threadIdx.x;
    if (tid < TOPK) { sF[tid] = feats[row * TOPK + tid]; sW[tid] = wts[row * TOPK + tid]; }
    __syncthreads();

    const int e0 = tid * 8;
    float acc[8];
#pragma unroll
    for (int j = 0; j < 8; ++j) acc[j] = bias[e0 + j];

    for (int k = 0; k < TOPK; ++k) {
        const float* lp = lookup + (size_t)sF[k] * EMBED + e0;
        const float  w  = sW[k];
        float4 a = *(const float4*)(lp);
        float4 b = *(const float4*)(lp + 4);
        acc[0] += w * a.x; acc[1] += w * a.y; acc[2] += w * a.z; acc[3] += w * a.w;
        acc[4] += w * b.x; acc[5] += w * b.y; acc[6] += w * b.z; acc[7] += w * b.w;
    }

    float ss = 0.0f;
#pragma unroll
    for (int j = 0; j < 8; ++j) ss += acc[j] * acc[j];
    red[tid] = ss;
    __syncthreads();
#pragma unroll
    for (int s = 128; s > 0; s >>= 1) {
        if (tid < s) red[tid] += red[tid + s];
        __syncthreads();
    }
    const float norm  = sqrtf(red[0]);
    const float scale = 1.0f / fmaxf(norm, 1e-12f);

    float* op = out + (size_t)row * EMBED + e0;
#pragma unroll
    for (int j = 0; j < 8; ++j) op[j] = acc[j] * scale;
}

// ---------------------------------------------------------------------------
// 6) usage update: new = (last + BATCH) * (1 - min(flag,1))
// ---------------------------------------------------------------------------
__global__ void usage_kernel(const int* __restrict__ last,
                             const int* __restrict__ flags,
                             int* __restrict__ outUsage) {
    int f = blockIdx.x * blockDim.x + threadIdx.x;
    if (f < NFEAT) outUsage[f] = flags[f] ? 0 : (last[f] + BATCH);
}

// ---------------------------------------------------------------------------
// launcher
// ---------------------------------------------------------------------------
extern "C" void kernel_launch(void* const* d_in, const int* in_sizes, int n_in,
                              void* d_out, int out_size, void* d_ws, size_t ws_size,
                              hipStream_t stream) {
    const float* embed      = (const float*)d_in[0];   // [BATCH x EMBED]
    const int*   last_usage = (const int*)  d_in[1];   // [NFEAT]
    const float* bias       = (const float*)d_in[2];   // [EMBED]
    const float* enc_weight = (const float*)d_in[3];   // [NFEAT x EMBED]
    const float* lookup     = (const float*)d_in[4];   // [NFEAT x EMBED]

    float* out_embed = (float*)d_out;                              // [BATCH x EMBED]
    int*   out_usage = (int*)(out_embed + (size_t)BATCH * EMBED);  // [NFEAT] int32 tail

    // workspace layout
    char* ws = (char*)d_ws;
    const size_t xbBytes = (size_t)BATCH * EMBED * sizeof(unsigned short); // 16 MB
    const size_t wbBytes = (size_t)NFEAT * EMBED * sizeof(unsigned short); // 128 MB
    const size_t pBytes  = (size_t)BATCH * NFEAT * sizeof(float);          // 512 MB
    const size_t fBytes  = (size_t)BATCH * TOPK  * sizeof(int);            // 1 MB
    unsigned short* Xb    = (unsigned short*)(ws);
    unsigned short* Wb    = (unsigned short*)(ws + xbBytes);
    float*          P     = (float*)(ws + xbBytes + wbBytes);
    int*            feats = (int*)  (ws + xbBytes + wbBytes + pBytes);
    float*          wts   = (float*)(ws + xbBytes + wbBytes + pBytes + fBytes);
    int*            flags = (int*)  (ws + xbBytes + wbBytes + pBytes + 2 * fBytes);

    zero_flags_kernel<<<NFEAT / 256, 256, 0, stream>>>(flags);
    convert_x_kernel<<<(BATCH * EMBED) / 256, 256, 0, stream>>>(embed, bias, Xb);
    convert_w_kernel<<<(NFEAT * EMBED) / 256, 256, 0, stream>>>(enc_weight, Wb);

    dim3 gGrid(NFEAT / BN, BATCH / BM);   // 256 x 32
    gemm_bf16_kernel<<<gGrid, 256, 0, stream>>>(Xb, Wb, P);

    topk_kernel<<<BATCH, 256, 0, stream>>>(P, feats, wts, flags);
    decode_norm_kernel<<<BATCH, 256, 0, stream>>>(feats, wts, lookup, bias, out_embed);
    usage_kernel<<<NFEAT / 256, 256, 0, stream>>>(last_usage, flags, out_usage);
}
